// EGNNLayer_23450521436535
// MI455X (gfx1250) — compile-verified, matrix operand order checked by reading, not run on hardware
//
#include <hip/hip_runtime.h>
#include <hip/hip_bf16.h>
#include <math.h>

// ---------------------------------------------------------------------------
// EGNN layer, fully fused, WMMA f16 path for MI455X (gfx1250, wave32).
//   B=4, N=512, H=64, MSG=64, NUM_RBFS=16, CUTOFF=5, LN_EPS=1e-5
// One wave per node i. Per 16-j tile: edge-MLP as 16x160 @ 160x64 ->
// 16x64 @ 64x64 -> 16x64 @ 64x64 (m_ij) -> 16x64 @ 64x64 (coord gate),
// all on v_wmma_f32_16x16x32_f16 with LDS-staged operands.
// GELU uses a branchless erf approximation (A&S 7.1.26, |err|<=1.5e-7)
// to keep the VALU side ~2-3x cheaper than libm erff and EXEC untouched.
// ---------------------------------------------------------------------------

typedef __attribute__((ext_vector_type(16))) _Float16 v16h;
typedef __attribute__((ext_vector_type(8)))  _Float16 v8h;
typedef __attribute__((ext_vector_type(8)))  float    v8f;

#define Bsz 4
#define Nn  512
#define Hh  64
#define KPAD 160   // 64 hi + 64 hj + 16 rbf + 16 zero pad

union V16u { v16h v; v8h h[2]; };

__device__ __forceinline__ v16h ldAB(const _Float16* p, int sel /*halves: 0 or 8*/) {
    V16u u;
    u.h[0] = *(const v8h*)(p + sel);        // 16B  -> ds_load_b128
    u.h[1] = *(const v8h*)(p + sel + 16);   // +32B -> ds_load_b128
    return u.v;
}

__device__ __forceinline__ v8f wmma_f16(v16h a, v16h b, v8f c) {
    return __builtin_amdgcn_wmma_f32_16x16x32_f16(false, a, false, b, (short)0, c,
                                                  false, false);
}

// GELU(v) = 0.5 v (1 + erf(v/sqrt(2))), branchless erf (A&S 7.1.26).
__device__ __forceinline__ float gelu_exact(float v) {
    float z  = v * 0.70710678118654752f;
    float az = __builtin_fabsf(z);
    float t  = __builtin_amdgcn_rcpf(__builtin_fmaf(0.3275911f, az, 1.0f));
    float p  = __builtin_fmaf(1.061405429f, t, -1.453152027f);
    p = __builtin_fmaf(p, t, 1.421413741f);
    p = __builtin_fmaf(p, t, -0.284496736f);
    p = __builtin_fmaf(p, t, 0.254829592f);
    p = p * t;
    float e  = __expf(-az * az);                 // v_exp_f32 path, branchless
    float er = __builtin_fmaf(-p, e, 1.0f);      // erf(|z|)
    er = __builtin_copysignf(er, z);
    return 0.5f * v * (1.0f + er);
}

__device__ __forceinline__ unsigned pack2h(float a, float b) {
    union { _Float16 h[2]; unsigned u; } q;
    q.h[0] = (_Float16)a; q.h[1] = (_Float16)b;
    return q.u;
}

__global__ void __launch_bounds__(256)
egnn_fused_kernel(const float* __restrict__ x, const float* __restrict__ h,
                  const unsigned char* __restrict__ node_mask,
                  const float* __restrict__ e_w1, const float* __restrict__ e_b1,
                  const float* __restrict__ e_w2, const float* __restrict__ e_b2,
                  const float* __restrict__ e_w3, const float* __restrict__ e_b3,
                  const float* __restrict__ n_w1, const float* __restrict__ n_b1,
                  const float* __restrict__ n_w2, const float* __restrict__ n_b2,
                  const float* __restrict__ n_w3, const float* __restrict__ n_b3,
                  const float* __restrict__ c_w1, const float* __restrict__ c_b1,
                  const float* __restrict__ c_w2, const float* __restrict__ c_b2,
                  const float* __restrict__ ln_g, const float* __restrict__ ln_b,
                  float* __restrict__ out_x, float* __restrict__ out_h)
{
    extern __shared__ __align__(16) char smem_raw[];
    _Float16* sm  = (_Float16*)smem_raw;
    _Float16* Bt1 = sm;               // e_w1^T padded: 64 x 160 halves
    _Float16* Bt2 = sm + 10240;       // e_w2^T: 64 x 64
    _Float16* Bt3 = sm + 14336;       // e_w3^T: 64 x 64
    _Float16* Btc = sm + 18432;       // c_w1^T: 64 x 64

    const int tid  = threadIdx.x;
    const int wave = tid >> 5;
    const int lane = tid & 31;

    // ---- cooperative weight preload (transpose + f16 convert, zero-pad K) ----
    for (int idx = tid; idx < 64 * KPAD; idx += 256) {
        int n = idx / KPAD, k = idx % KPAD;
        Bt1[n * KPAD + k] = (k < 144) ? (_Float16)e_w1[k * 64 + n] : (_Float16)0.0f;
    }
    for (int idx = tid; idx < 64 * 64; idx += 256) {
        int n = idx >> 6, k = idx & 63;
        Bt2[n * 64 + k] = (_Float16)e_w2[k * 64 + n];
        Bt3[n * 64 + k] = (_Float16)e_w3[k * 64 + n];
        Btc[n * 64 + k] = (_Float16)c_w1[k * 64 + n];
    }
    __syncthreads();

    // per-wave staging: A1 (16x160 f16), Astg (16x64 f16), float scratch
    _Float16* wb      = sm + 22528 + wave * 4032;
    _Float16* A1      = wb;                 // 2560 halves
    _Float16* Astg    = wb + 2560;          // 1024 halves
    float*    fbuf    = (float*)(wb + 3584);
    float*    maskbuf = fbuf;               // 16
    float*    gbuf    = fbuf + 16;          // 16
    float*    vecIn   = fbuf + 32;          // 128
    float*    vecU    = fbuf + 160;         // 64

    const int b = blockIdx.x >> 6;          // N/8 = 64 blocks per batch
    const int i = (blockIdx.x & 63) * 8 + wave;
    const int ln16 = lane & 15;
    const int hi16 = lane >> 4;             // 0 or 1
    const int rsel = hi16 * 8;              // operand half-select in halves

    // per-lane column biases (column = nt*16 + ln16)
    float eb1v[4], eb2v[4], eb3v[4], cb1v[4], cw2v[4];
#pragma unroll
    for (int nt = 0; nt < 4; nt++) {
        int c = nt * 16 + ln16;
        eb1v[nt] = e_b1[c]; eb2v[nt] = e_b2[c]; eb3v[nt] = e_b3[c];
        cb1v[nt] = c_b1[c]; cw2v[nt] = c_w2[c];
    }
    const float cb2 = c_b2[0];

    const float xi0 = x[(b * Nn + i) * 3 + 0];
    const float xi1 = x[(b * Nn + i) * 3 + 1];
    const float xi2 = x[(b * Nn + i) * 3 + 2];
    const bool  nmi = node_mask[b * Nn + i] != 0;

    // ---- fill h_i region of A1 (same across all 16 rows), zero K-pad ----
    {
        const float* hi = h + (size_t)(b * Nn + i) * Hh;
        float2 hv = *(const float2*)(hi + lane * 2);
        unsigned pv = pack2h(hv.x, hv.y);
#pragma unroll
        for (int r = 0; r < 16; r++)
            *(unsigned*)(A1 + r * KPAD + lane * 2) = pv;
        if (lane < 16) {
#pragma unroll
            for (int q = 0; q < 8; q++)
                *(unsigned*)(A1 + lane * KPAD + 144 + q * 2) = 0u;
        }
    }

    float macc[4] = {0.f, 0.f, 0.f, 0.f};             // m_i partials
    float del0 = 0.f, del1 = 0.f, del2 = 0.f, cnt = 0.f;
    float dxr0 = 0.f, dxr1 = 0.f, dxr2 = 0.f, maskf = 0.f;

    for (int j0 = 0; j0 < Nn; j0 += 16) {
        // prefetch next tile's h rows into cache while this tile computes
        if (j0 + 16 < Nn) {
            const float* pf = h + (size_t)(b * Nn + j0 + 16 + (lane >> 1)) * Hh
                                + (lane & 1) * 32;
            __builtin_prefetch(pf, 0, 1);
        }
        // ---- build h_j region: 16 rows x 64 cols as f16 ----
#pragma unroll 4
        for (int t = 0; t < 16; t++) {
            int p   = lane + 32 * t;      // 0..511 float2 pairs
            int row = p >> 5;
            int cp  = p & 31;
            float2 hv = *(const float2*)(h + (size_t)(b * Nn + j0 + row) * Hh + cp * 2);
            *(unsigned*)(A1 + row * KPAD + 64 + cp * 2) = pack2h(hv.x, hv.y);
        }
        // ---- rbf + geometry per row (lanes 0..15) ----
        if (lane < 16) {
            int j = j0 + lane;
            float xj0 = x[(b * Nn + j) * 3 + 0];
            float xj1 = x[(b * Nn + j) * 3 + 1];
            float xj2 = x[(b * Nn + j) * 3 + 2];
            dxr0 = xi0 - xj0; dxr1 = xi1 - xj1; dxr2 = xi2 - xj2;
            float d2   = dxr0 * dxr0 + dxr1 * dxr1 + dxr2 * dxr2;
            float dist = sqrtf(d2 + 1e-8f);
            bool  mk   = (dist <= 5.0f) && (j != i) && nmi &&
                         (node_mask[b * Nn + j] != 0);
            maskf = mk ? 1.0f : 0.0f;
            maskbuf[lane] = maskf;
            unsigned* rp = (unsigned*)(A1 + lane * KPAD + 128);
#pragma unroll
            for (int k = 0; k < 16; k += 2) {
                float t0 = (dist - (float)k       * (5.0f / 15.0f)) * 3.0f;
                float t1 = (dist - (float)(k + 1) * (5.0f / 15.0f)) * 3.0f;
                rp[k >> 1] = pack2h(__expf(-0.5f * t0 * t0),
                                    __expf(-0.5f * t1 * t1));
            }
        }

        // ================= GEMM1: 16x160 @ 160x64, bias+GELU =================
        v16h a1[5];
        const _Float16* arow = A1 + ln16 * KPAD;
#pragma unroll
        for (int ks = 0; ks < 5; ks++) a1[ks] = ldAB(arow + ks * 32, rsel);
#pragma unroll
        for (int nt = 0; nt < 4; nt++) {
            const _Float16* brow = Bt1 + (nt * 16 + ln16) * KPAD;
            v8f c = {};
#pragma unroll
            for (int ks = 0; ks < 5; ks++)
                c = wmma_f16(a1[ks], ldAB(brow + ks * 32, rsel), c);
#pragma unroll
            for (int r = 0; r < 8; r++) {
                float v = gelu_exact(c[r] + eb1v[nt]);
                Astg[(r + hi16 * 8) * 64 + nt * 16 + ln16] = (_Float16)v;
            }
        }

        // ================= GEMM2: 16x64 @ 64x64, bias+GELU ===================
        v16h a2[2];
        const _Float16* srow = Astg + ln16 * 64;
#pragma unroll
        for (int ks = 0; ks < 2; ks++) a2[ks] = ldAB(srow + ks * 32, rsel);
#pragma unroll
        for (int nt = 0; nt < 4; nt++) {
            const _Float16* brow = Bt2 + (nt * 16 + ln16) * 64;
            v8f c = {};
#pragma unroll
            for (int ks = 0; ks < 2; ks++)
                c = wmma_f16(a2[ks], ldAB(brow + ks * 32, rsel), c);
#pragma unroll
            for (int r = 0; r < 8; r++) {
                float v = gelu_exact(c[r] + eb2v[nt]);
                Astg[(r + hi16 * 8) * 64 + nt * 16 + ln16] = (_Float16)v;
            }
        }

        // ================= GEMM3: m_ij = 16x64 @ 64x64, bias+mask ============
        float mk[8];
#pragma unroll
        for (int r = 0; r < 8; r++) mk[r] = maskbuf[hi16 * 8 + r];

        v16h a3[2];
#pragma unroll
        for (int ks = 0; ks < 2; ks++) a3[ks] = ldAB(srow + ks * 32, rsel);
#pragma unroll
        for (int nt = 0; nt < 4; nt++) {
            const _Float16* brow = Bt3 + (nt * 16 + ln16) * 64;
            v8f c = {};
#pragma unroll
            for (int ks = 0; ks < 2; ks++)
                c = wmma_f16(a3[ks], ldAB(brow + ks * 32, rsel), c);
            float acc = 0.f;
#pragma unroll
            for (int r = 0; r < 8; r++) {
                float v = (c[r] + eb3v[nt]) * mk[r];   // masked m_ij
                acc += v;
                Astg[(r + hi16 * 8) * 64 + nt * 16 + ln16] = (_Float16)v;
            }
            macc[nt] += acc;
        }

        // ============ GEMM4: gate g = GELU(m_ij @ c_w1 + b) @ c_w2 ===========
        v16h a4[2];
#pragma unroll
        for (int ks = 0; ks < 2; ks++) a4[ks] = ldAB(srow + ks * 32, rsel);
        float part[8] = {0.f,0.f,0.f,0.f,0.f,0.f,0.f,0.f};
#pragma unroll
        for (int nt = 0; nt < 4; nt++) {
            const _Float16* brow = Btc + (nt * 16 + ln16) * 64;
            v8f c = {};
#pragma unroll
            for (int ks = 0; ks < 2; ks++)
                c = wmma_f16(a4[ks], ldAB(brow + ks * 32, rsel), c);
#pragma unroll
            for (int r = 0; r < 8; r++)
                part[r] += gelu_exact(c[r] + cb1v[nt]) * cw2v[nt];
        }
        // reduce over the 16 lanes of each half (N dimension)
#pragma unroll
        for (int r = 0; r < 8; r++) {
            float s = part[r];
            s += __shfl_xor(s, 1); s += __shfl_xor(s, 2);
            s += __shfl_xor(s, 4); s += __shfl_xor(s, 8);
            part[r] = s;
        }
        if (ln16 == 0) {
#pragma unroll
            for (int r = 0; r < 8; r++)
                gbuf[hi16 * 8 + r] = (part[r] + cb2) * mk[r];
        }
        if (lane < 16) {
            float g = gbuf[lane];
            del0 += dxr0 * g; del1 += dxr1 * g; del2 += dxr2 * g;
            cnt  += maskf;
        }
    } // j-tile loop

    // ---- combine m_i halves (rows 0-7 in lanes<16, rows 8-15 in lanes>=16) --
#pragma unroll
    for (int nt = 0; nt < 4; nt++) macc[nt] += __shfl_xor(macc[nt], 16);

    // ======================= node MLP + residual + LN =======================
    const size_t hb = (size_t)(b * Nn + i) * Hh;
    vecIn[lane]      = h[hb + lane];
    vecIn[lane + 32] = h[hb + lane + 32];
    if (lane < 16) {
#pragma unroll
        for (int nt = 0; nt < 4; nt++) vecIn[64 + nt * 16 + lane] = macc[nt];
    }
    float o0 = n_b1[lane], o1 = n_b1[lane + 32];
    for (int k = 0; k < 128; k++) {
        float xv = vecIn[k];
        o0 += xv * n_w1[k * 64 + lane];
        o1 += xv * n_w1[k * 64 + lane + 32];
    }
    vecU[lane] = gelu_exact(o0); vecU[lane + 32] = gelu_exact(o1);

    o0 = n_b2[lane]; o1 = n_b2[lane + 32];
    for (int k = 0; k < 64; k++) {
        float xv = vecU[k];
        o0 += xv * n_w2[k * 64 + lane];
        o1 += xv * n_w2[k * 64 + lane + 32];
    }
    vecIn[lane] = gelu_exact(o0); vecIn[lane + 32] = gelu_exact(o1);

    o0 = n_b3[lane]; o1 = n_b3[lane + 32];
    for (int k = 0; k < 64; k++) {
        float xv = vecIn[k];
        o0 += xv * n_w3[k * 64 + lane];
        o1 += xv * n_w3[k * 64 + lane + 32];
    }
    float hr0 = h[hb + lane] + o0;
    float hr1 = h[hb + lane + 32] + o1;

    float s = hr0 + hr1;
    s += __shfl_xor(s, 1); s += __shfl_xor(s, 2); s += __shfl_xor(s, 4);
    s += __shfl_xor(s, 8); s += __shfl_xor(s, 16);
    float mu = s * (1.0f / 64.0f);
    float q0 = hr0 - mu, q1 = hr1 - mu;
    float v = q0 * q0 + q1 * q1;
    v += __shfl_xor(v, 1); v += __shfl_xor(v, 2); v += __shfl_xor(v, 4);
    v += __shfl_xor(v, 8); v += __shfl_xor(v, 16);
    float inv = rsqrtf(v * (1.0f / 64.0f) + 1e-5f);
    out_h[hb + lane]      = q0 * inv * ln_g[lane]      + ln_b[lane];
    out_h[hb + lane + 32] = q1 * inv * ln_g[lane + 32] + ln_b[lane + 32];

    // ======================= coordinate update ==============================
    float d0 = del0, d1 = del1, d2 = del2, cc = cnt;
#pragma unroll
    for (int m = 1; m <= 16; m <<= 1) {
        d0 += __shfl_xor(d0, m); d1 += __shfl_xor(d1, m);
        d2 += __shfl_xor(d2, m); cc += __shfl_xor(cc, m);
    }
    if (lane == 0) {
        float dn = fmaxf(cc, 1.0f);
        size_t xb = (size_t)(b * Nn + i) * 3;
        out_x[xb + 0] = x[xb + 0] + d0 / dn;
        out_x[xb + 1] = x[xb + 1] + d1 / dn;
        out_x[xb + 2] = x[xb + 2] + d2 / dn;
    }
}

extern "C" void kernel_launch(void* const* d_in, const int* in_sizes, int n_in,
                              void* d_out, int out_size, void* d_ws, size_t ws_size,
                              hipStream_t stream) {
    (void)in_sizes; (void)n_in; (void)out_size; (void)d_ws; (void)ws_size;
    const float* x    = (const float*)d_in[0];
    const float* h    = (const float*)d_in[1];
    const unsigned char* nm = (const unsigned char*)d_in[2];
    const float* e_w1 = (const float*)d_in[3];
    const float* e_b1 = (const float*)d_in[4];
    const float* e_w2 = (const float*)d_in[5];
    const float* e_b2 = (const float*)d_in[6];
    const float* e_w3 = (const float*)d_in[7];
    const float* e_b3 = (const float*)d_in[8];
    const float* n_w1 = (const float*)d_in[9];
    const float* n_b1 = (const float*)d_in[10];
    const float* n_w2 = (const float*)d_in[11];
    const float* n_b2 = (const float*)d_in[12];
    const float* n_w3 = (const float*)d_in[13];
    const float* n_b3 = (const float*)d_in[14];
    const float* c_w1 = (const float*)d_in[15];
    const float* c_b1 = (const float*)d_in[16];
    const float* c_w2 = (const float*)d_in[17];
    const float* c_b2 = (const float*)d_in[18];
    const float* ln_g = (const float*)d_in[19];
    const float* ln_b = (const float*)d_in[20];

    float* out_x = (float*)d_out;                 // 4*512*3
    float* out_h = out_x + Bsz * Nn * 3;          // 4*512*64

    const size_t shmem = 109568;                  // weights + 8 wave stages
    dim3 grid(Bsz * (Nn / 8));                    // 256 blocks
    dim3 block(256);                              // 8 waves, one node i each
    egnn_fused_kernel<<<grid, block, shmem, stream>>>(
        x, h, nm, e_w1, e_b1, e_w2, e_b2, e_w3, e_b3,
        n_w1, n_b1, n_w2, n_b2, n_w3, n_b3,
        c_w1, c_b1, c_w2, c_b2, ln_g, ln_b, out_x, out_h);
}
